// QwenImageTransformer2DModel_56796647523111
// MI455X (gfx1250) — matrix operand order repaired and checked by reading, not compiled, not run on hardware
//
#include <hip/hip_runtime.h>
#include <cstdint>
#include <cstddef>

#define DIMX   3072
#define MLPX   12288
#define S_TXT  512
#define S_IMG  2048
#define S_ALL  2560
#define HEADSX 24
#define DHX    128
#define EPSX   1e-6f

typedef __attribute__((ext_vector_type(16))) __bf16 v16bf;
typedef __attribute__((ext_vector_type(8)))  float  v8f;
typedef int v4i_ __attribute__((vector_size(16)));   // matches builtin param type

union Frag {
    v16bf v;
    uint32_t u[8];
};

// ---------------------------------------------------------------------------
// CDNA5 async global->LDS copy (GLOBAL_LOAD_ASYNC_TO_LDS_B128, ASYNCcnt).
// Guarded: falls back to plain copy if toolchain lacks the builtin.
// Probe result: param0 is 'int __vector_size__(16) __device__ *' (global AS).
// ---------------------------------------------------------------------------
#if defined(__has_builtin)
#if __has_builtin(__builtin_amdgcn_global_load_async_to_lds_b128) && \
    __has_builtin(__builtin_amdgcn_s_wait_asynccnt)
#define USE_ASYNC_LDS 1
#endif
#endif
#ifndef USE_ASYNC_LDS
#define USE_ASYNC_LDS 0
#endif

__device__ __forceinline__ void async_copy_b128(const uint32_t* g, uint32_t* l) {
#if USE_ASYNC_LDS
    __builtin_amdgcn_global_load_async_to_lds_b128(
        (__attribute__((address_space(1))) v4i_*)g,
        (__attribute__((address_space(3))) v4i_*)l,
        /*offset=*/0, /*cpol=*/0);
#else
    l[0] = g[0]; l[1] = g[1]; l[2] = g[2]; l[3] = g[3];
#endif
}

__device__ __forceinline__ void async_wait_all() {
#if USE_ASYNC_LDS
    __builtin_amdgcn_s_wait_asynccnt(0);
#endif
}

__device__ __forceinline__ uint16_t f2bf(float f) {
    union { float f; uint32_t u; } c; c.f = f;
    uint32_t u = c.u;
    uint32_t r = u + 0x7FFFu + ((u >> 16) & 1u);   // round-to-nearest-even
    return (uint16_t)(r >> 16);
}

__device__ __forceinline__ float gelu_tanh(float x) {
    float x3 = x * x * x;
    return 0.5f * x * (1.0f + tanhf(0.7978845608028654f * (x + 0.044715f * x3)));
}

__device__ __forceinline__ v8f wmma_bf16(const Frag& a, const Frag& b, v8f c) {
    return __builtin_amdgcn_wmma_f32_16x16x32_bf16(
        /*neg_a=*/false, a.v, /*neg_b=*/false, b.v,
        /*c_mod=*/(short)0, c, /*reuse_a=*/false, /*reuse_b=*/false);
}

// ---------------------------------------------------------------------------
// small elementwise kernels
// ---------------------------------------------------------------------------
__global__ void silu_kernel(const float* __restrict__ x, float* __restrict__ o, int n) {
    int i = blockIdx.x * 256 + threadIdx.x;
    if (i < n) { float v = x[i]; o[i] = v / (1.0f + __expf(-v)); }
}

__global__ void cvt_bf16_kernel(const float* __restrict__ x, uint16_t* __restrict__ o, size_t n) {
    size_t i = (size_t)blockIdx.x * 256 + threadIdx.x;
    size_t stride = (size_t)gridDim.x * 256;
    for (; i < n; i += stride) o[i] = f2bf(x[i]);
}

__global__ void mask_kernel(const float* __restrict__ m, float* __restrict__ o) {
    int i = blockIdx.x * 256 + threadIdx.x;
    if (i < S_ALL) o[i] = (i < S_TXT) ? m[i] : 0.0f;
}

// mod = silu(temb) @ W + b, for img and txt  (out: [6*DIMX] each)
__global__ void modvec_kernel(const float* __restrict__ t,
                              const float* __restrict__ Wi, const float* __restrict__ Bi,
                              const float* __restrict__ Wt, const float* __restrict__ Bt,
                              float* __restrict__ Mi, float* __restrict__ Mt) {
    int o = blockIdx.x * 256 + threadIdx.x;        // 0..36863
    int mat = o / (6 * DIMX);
    int col = o % (6 * DIMX);
    const float* W = mat ? Wt : Wi;
    const float* B = mat ? Bt : Bi;
    float* out     = mat ? Mt : Mi;
    float acc = B[col];
    for (int k = 0; k < DIMX; ++k)
        acc += t[k] * W[(size_t)k * (6 * DIMX) + col];
    out[col] = acc;
}

// LayerNorm (no affine) + modulate -> bf16.  rows<S_TXT come from xt, else xi.
__global__ __launch_bounds__(256)
void ln_mod_kernel(const float* __restrict__ xt, const float* __restrict__ xi,
                   const float* __restrict__ modt, const float* __restrict__ modi,
                   uint16_t* __restrict__ out) {
    __shared__ float red[16];
    int row = blockIdx.x;
    const float* x   = (row < S_TXT) ? (xt + (size_t)row * DIMX)
                                     : (xi + (size_t)(row - S_TXT) * DIMX);
    const float* mod = (row < S_TXT) ? modt : modi;
    int tid = threadIdx.x;
    float s = 0.0f, ss = 0.0f;
    for (int c = tid; c < DIMX; c += 256) { float v = x[c]; s += v; ss += v * v; }
    for (int o = 1; o < 32; o <<= 1) { s += __shfl_xor(s, o, 32); ss += __shfl_xor(ss, o, 32); }
    int wid = tid >> 5, lane = tid & 31;
    if (lane == 0) { red[wid] = s; red[8 + wid] = ss; }
    __syncthreads();
    float st = 0.0f, sst = 0.0f;
#pragma unroll
    for (int w = 0; w < 8; ++w) { st += red[w]; sst += red[8 + w]; }
    float mean = st * (1.0f / DIMX);
    float var  = sst * (1.0f / DIMX) - mean * mean;
    float inv  = rsqrtf(var + EPSX);
    for (int c = tid; c < DIMX; c += 256) {
        float xh = (x[c] - mean) * inv;
        float v  = mod[c] + xh * (1.0f + mod[DIMX + c]);
        out[(size_t)row * DIMX + c] = f2bf(v);
    }
}

// RMSNorm over head dim + RoPE -> bf16. one block per (s, head), 128 threads.
__global__ __launch_bounds__(128)
void rmsrope_kernel(const float* __restrict__ xf, const float* __restrict__ wt,
                    const float* __restrict__ wi, const float* __restrict__ freqs,
                    uint16_t* __restrict__ out) {
    __shared__ float red[4];
    int s = blockIdx.x / HEADSX, head = blockIdx.x % HEADSX;
    int tid = threadIdx.x;
    size_t base = (size_t)s * DIMX + head * DHX;
    float v = xf[base + tid];
    float sq = v * v;
    for (int o = 1; o < 32; o <<= 1) sq += __shfl_xor(sq, o, 32);
    int wid = tid >> 5, lane = tid & 31;
    if (lane == 0) red[wid] = sq;
    __syncthreads();
    float tot = red[0] + red[1] + red[2] + red[3];
    float inv = rsqrtf(tot * (1.0f / DHX) + EPSX);
    float w = ((s < S_TXT) ? wt : wi)[tid];
    float xn = v * inv * w;
    float partner = __shfl_xor(xn, 1, 32);
    int pair = tid >> 1, r = tid & 1;
    float xe = r ? partner : xn;
    float xo = r ? xn : partner;
    const float* fr = freqs + ((size_t)s * 64 + pair) * 4 + r * 2;
    out[base + tid] = f2bf(fr[0] * xe + fr[1] * xo);
}

// out[row][c] = base[row][c] + gate[row-domain][c] * x[row][c]
__global__ void gate_add_kernel(const float* __restrict__ bt, const float* __restrict__ bi,
                                const float* __restrict__ x,
                                const float* __restrict__ gt, const float* __restrict__ gi,
                                float* __restrict__ out, size_t n) {
    size_t i = (size_t)blockIdx.x * 256 + threadIdx.x;
    if (i >= n) return;
    size_t row = i / DIMX;
    int c = (int)(i % DIMX);
    float base = (row < S_TXT) ? bt[row * DIMX + c] : bi[(row - S_TXT) * DIMX + c];
    float g = ((row < S_TXT) ? gt : gi)[c];
    out[i] = base + g * x[i];
}

// ---------------------------------------------------------------------------
// bf16 WMMA GEMM: C[M,N] = A[M,K] (bf16, rowmajor) * B[K,N] (bf16, rowmajor) + bias
// mode 0: fp32 out; mode 1: gelu -> bf16 out; mode 2: bf16 out
// tile 128x128x32, 256 threads = 8 waves (4 along M x 2 along N), 32x64 per wave
// A tile staged via GLOBAL_LOAD_ASYNC_TO_LDS_B128 when available.
// ---------------------------------------------------------------------------
__global__ __launch_bounds__(256)
void gemm_bf16_kernel(const uint16_t* __restrict__ A, const uint16_t* __restrict__ B,
                      const float* __restrict__ bias, float* __restrict__ outF,
                      uint16_t* __restrict__ outH, int M, int N, int K, int mode) {
    __shared__ uint32_t lA[128 * 20];     // A tile [128][32] bf16, row stride 20 u32 (16B aligned rows)
    __shared__ uint16_t lB[128 * 34];     // B tile transposed [n=128][k=32], stride 34 u16

    const int tid  = threadIdx.x;
    const int lane = tid & 31;
    const int wid  = tid >> 5;
    const int wm   = wid & 3;             // wave M slot (0..3) -> rows wm*32
    const int wn   = wid >> 2;            // wave N slot (0..1) -> cols wn*64
    const int h    = lane >> 4;           // half-wave select
    const int n16  = lane & 15;
    const int m0   = blockIdx.y * 128;
    const int n0   = blockIdx.x * 128;

    const uint32_t* gA = (const uint32_t*)A;
    const uint32_t* gB = (const uint32_t*)B;
    const int Ku = K >> 1;
    const int Nu = N >> 1;

    const int ar = tid >> 1, ah = tid & 1;     // A loader: row, 16-bf16 half
    const int br = tid >> 3, bs = tid & 7;     // B loader: k row, 16-col segment

    v8f acc[2][4];
#pragma unroll
    for (int mt = 0; mt < 2; ++mt)
#pragma unroll
        for (int nt = 0; nt < 4; ++nt)
#pragma unroll
            for (int r = 0; r < 8; ++r) acc[mt][nt][r] = 0.0f;

    for (int k0 = 0; k0 < K; k0 += 32) {
        __syncthreads();
        {   // stage A: 2x async B128 per thread (global -> LDS, no VGPR round trip)
            const uint32_t* src = gA + (size_t)(m0 + ar) * Ku + (k0 >> 1) + ah * 8;
            uint32_t* dst = lA + ar * 20 + ah * 8;
            async_copy_b128(src,     dst);
            async_copy_b128(src + 4, dst + 4);
        }
        {   // stage B transposed into LDS (transpose -> synchronous path)
            const uint32_t* src = gB + (size_t)(k0 + br) * Nu + (n0 >> 1) + bs * 8;
#pragma unroll
            for (int j = 0; j < 8; ++j) {
                uint32_t wv = src[j];
                int c = bs * 16 + 2 * j;
                lB[(c + 0) * 34 + br] = (uint16_t)(wv & 0xFFFFu);
                lB[(c + 1) * 34 + br] = (uint16_t)(wv >> 16);
            }
        }
        if (k0 + 32 < K) {   // prefetch next B tile -> global_prefetch_b8
            __builtin_prefetch(gB + (size_t)(k0 + 32 + br) * Nu + (n0 >> 1) + bs * 8, 0, 1);
        }
        async_wait_all();
        __syncthreads();

        Frag a[2], b[4];
#pragma unroll
        for (int mt = 0; mt < 2; ++mt) {
            int row = wm * 32 + mt * 16 + n16;
#pragma unroll
            for (int v = 0; v < 8; ++v) {
                int kk = 16 * h + 2 * v;
                a[mt].u[v] = lA[row * 20 + (kk >> 1)];
            }
        }
        const uint32_t* lBu = (const uint32_t*)lB;
#pragma unroll
        for (int nt = 0; nt < 4; ++nt) {
            int col = wn * 64 + nt * 16 + n16;
#pragma unroll
            for (int v = 0; v < 8; ++v)
                b[nt].u[v] = lBu[col * 17 + 8 * h + v];   // k pair = 8h+v  (K=16h+2v)
        }
#pragma unroll
        for (int mt = 0; mt < 2; ++mt)
#pragma unroll
            for (int nt = 0; nt < 4; ++nt)
                acc[mt][nt] = wmma_bf16(a[mt], b[nt], acc[mt][nt]);
    }

#pragma unroll
    for (int mt = 0; mt < 2; ++mt) {
#pragma unroll
        for (int nt = 0; nt < 4; ++nt) {
            int col = n0 + wn * 64 + nt * 16 + n16;
            float bv = bias[col];
#pragma unroll
            for (int r = 0; r < 8; ++r) {
                int row = m0 + wm * 32 + mt * 16 + 8 * h + r;
                float val = acc[mt][nt][r] + bv;
                size_t off = (size_t)row * N + col;
                if (mode == 0)      outF[off] = val;
                else if (mode == 1) outH[off] = f2bf(gelu_tanh(val));
                else                outH[off] = f2bf(val);
            }
        }
    }
}

// ---------------------------------------------------------------------------
// Flash attention with WMMA. grid = (S_ALL/128, HEADSX), 256 threads = 8 waves.
// Each wave owns 16 q rows; loop over 32-key blocks with online softmax.
// Q/K/V are bf16 [S_ALL][DIMX] (head at col head*DHX). O is bf16 same layout.
// K tile staged via async-to-LDS; V tile transposed synchronously.
// ---------------------------------------------------------------------------
__global__ __launch_bounds__(256)
void flash_attn_kernel(const uint16_t* __restrict__ Q, const uint16_t* __restrict__ Km,
                       const uint16_t* __restrict__ Vm, const float* __restrict__ maskf,
                       uint16_t* __restrict__ O) {
    __shared__ uint32_t lK4[32 * 68];      // [key][dh] bf16, row stride 68 u32 (16B aligned rows)
    __shared__ uint16_t lV[128 * 34];      // transposed [dh][key] stride 34
    __shared__ uint16_t lP[8][16 * 34];    // per-wave P tile [16][32] stride 34

    const int tid  = threadIdx.x;
    const int lane = tid & 31;
    const int wid  = tid >> 5;
    const int h    = lane >> 4;
    const int n16  = lane & 15;
    const int head = blockIdx.y;
    const int q0   = blockIdx.x * 128;
    const int qrow = q0 + wid * 16;
    const float scale = 0.08838834764831845f;   // 1/sqrt(128)

    const uint32_t* gQ = (const uint32_t*)Q;
    const uint32_t* gK = (const uint32_t*)Km;
    const uint32_t* gV = (const uint32_t*)Vm;

    // preload Q fragments for the 4 K-dim steps
    Frag qa[4];
    {
        size_t base = (size_t)(qrow + n16) * (DIMX / 2) + head * (DHX / 2);
#pragma unroll
        for (int kk = 0; kk < 4; ++kk)
#pragma unroll
            for (int v = 0; v < 8; ++v) {
                int kd = kk * 32 + 16 * h + 2 * v;
                qa[kk].u[v] = gQ[base + (kd >> 1)];
            }
    }

    float mrow[8], lrow[8];
    v8f o_acc[8];
#pragma unroll
    for (int r = 0; r < 8; ++r) { mrow[r] = -1e30f; lrow[r] = 0.0f; }
#pragma unroll
    for (int j = 0; j < 8; ++j)
#pragma unroll
        for (int r = 0; r < 8; ++r) o_acc[j][r] = 0.0f;

    for (int kb = 0; kb < S_ALL / 32; ++kb) {
        const int key0 = kb * 32;
        __syncthreads();
        {   // stage K (async row-major copy) and V (transposed, sync)
            int key = tid >> 3, seg = tid & 7;
            const uint32_t* ks = gK + (size_t)(key0 + key) * (DIMX / 2) + head * (DHX / 2) + seg * 8;
            uint32_t* kd = lK4 + key * 68 + seg * 8;
            async_copy_b128(ks,     kd);
            async_copy_b128(ks + 4, kd + 4);
            const uint32_t* vs = gV + (size_t)(key0 + key) * (DIMX / 2) + head * (DHX / 2) + seg * 8;
#pragma unroll
            for (int j = 0; j < 8; ++j) {
                uint32_t wv = vs[j];
                int c = seg * 16 + 2 * j;
                lV[(c + 0) * 34 + key] = (uint16_t)(wv & 0xFFFFu);
                lV[(c + 1) * 34 + key] = (uint16_t)(wv >> 16);
            }
        }
        async_wait_all();
        __syncthreads();

        // S = Q K^T : two 16x16 D tiles over 4 k-steps
        v8f s0, s1;
#pragma unroll
        for (int r = 0; r < 8; ++r) { s0[r] = 0.0f; s1[r] = 0.0f; }
#pragma unroll
        for (int kk = 0; kk < 4; ++kk) {
            Frag kb0, kb1;
#pragma unroll
            for (int v = 0; v < 8; ++v) {
                kb0.u[v] = lK4[(0 * 16 + n16) * 68 + kk * 16 + 8 * h + v];
                kb1.u[v] = lK4[(1 * 16 + n16) * 68 + kk * 16 + 8 * h + v];
            }
            s0 = wmma_bf16(qa[kk], kb0, s0);
            s1 = wmma_bf16(qa[kk], kb1, s1);
        }

        float mk0 = maskf[key0 + n16];
        float mk1 = maskf[key0 + 16 + n16];
#pragma unroll
        for (int r = 0; r < 8; ++r) { s0[r] = s0[r] * scale + mk0; s1[r] = s1[r] * scale + mk1; }

        // online softmax: row max, alpha, P, row sum
        float mnew[8], alpha[8], lblk[8];
#pragma unroll
        for (int r = 0; r < 8; ++r) {
            float mb = fmaxf(s0[r], s1[r]);
#pragma unroll
            for (int o = 1; o < 16; o <<= 1) mb = fmaxf(mb, __shfl_xor(mb, o, 32));
            mnew[r] = fmaxf(mrow[r], mb);
            alpha[r] = __expf(mrow[r] - mnew[r]);
            mrow[r] = mnew[r];
        }
        uint16_t* myP = &lP[wid][0];
#pragma unroll
        for (int r = 0; r < 8; ++r) {
            float p0 = __expf(s0[r] - mnew[r]);
            float p1 = __expf(s1[r] - mnew[r]);
            lblk[r] = p0 + p1;
            myP[(8 * h + r) * 34 + n16]      = f2bf(p0);
            myP[(8 * h + r) * 34 + 16 + n16] = f2bf(p1);
        }
#pragma unroll
        for (int r = 0; r < 8; ++r) {
            float lb = lblk[r];
#pragma unroll
            for (int o = 1; o < 16; o <<= 1) lb += __shfl_xor(lb, o, 32);
            lrow[r] = lrow[r] * alpha[r] + lb;
        }
#pragma unroll
        for (int j = 0; j < 8; ++j)
#pragma unroll
            for (int r = 0; r < 8; ++r) o_acc[j][r] *= alpha[r];

        // O += P @ V   (P: 16x32 A-frag from wave-private LDS; V^T in LDS)
        Frag pa;
        {
            const uint32_t* myPu = (const uint32_t*)myP;
#pragma unroll
            for (int v = 0; v < 8; ++v) {
                int kd = 16 * h + 2 * v;
                pa.u[v] = myPu[n16 * 17 + (kd >> 1)];
            }
        }
        const uint32_t* lVu = (const uint32_t*)lV;
#pragma unroll
        for (int j = 0; j < 8; ++j) {
            Frag vbf;
            int dcol = j * 16 + n16;
#pragma unroll
            for (int v = 0; v < 8; ++v) vbf.u[v] = lVu[dcol * 17 + 8 * h + v];
            o_acc[j] = wmma_bf16(pa, vbf, o_acc[j]);
        }
    }

    // normalize and write bf16 [row][head*128 + dh]
#pragma unroll
    for (int j = 0; j < 8; ++j) {
        int col = head * DHX + j * 16 + n16;
#pragma unroll
        for (int r = 0; r < 8; ++r) {
            int row = qrow + 8 * h + r;
            O[(size_t)row * DIMX + col] = f2bf(o_acc[j][r] / lrow[r]);
        }
    }
}

// ---------------------------------------------------------------------------
// host side
// ---------------------------------------------------------------------------
static inline size_t alnup(size_t x) { return (x + 255) & ~(size_t)255; }

extern "C" void kernel_launch(void* const* d_in, const int* in_sizes, int n_in,
                              void* d_out, int out_size, void* d_ws, size_t ws_size,
                              hipStream_t stream) {
    (void)in_sizes; (void)n_in; (void)out_size; (void)ws_size;
    const float* hs        = (const float*)d_in[0];
    const float* ehs       = (const float*)d_in[1];
    const float* mask_in   = (const float*)d_in[2];
    const float* temb      = (const float*)d_in[3];
    const float* freqs     = (const float*)d_in[4];
    const float* img_mod_w = (const float*)d_in[5];
    const float* img_mod_b = (const float*)d_in[6];
    const float* txt_mod_w = (const float*)d_in[7];
    const float* txt_mod_b = (const float*)d_in[8];
    const float* to_q_w  = (const float*)d_in[9];   const float* to_q_b  = (const float*)d_in[10];
    const float* to_k_w  = (const float*)d_in[11];  const float* to_k_b  = (const float*)d_in[12];
    const float* to_v_w  = (const float*)d_in[13];  const float* to_v_b  = (const float*)d_in[14];
    const float* add_q_w = (const float*)d_in[15];  const float* add_q_b = (const float*)d_in[16];
    const float* add_k_w = (const float*)d_in[17];  const float* add_k_b = (const float*)d_in[18];
    const float* add_v_w = (const float*)d_in[19];  const float* add_v_b = (const float*)d_in[20];
    const float* norm_q_w       = (const float*)d_in[21];
    const float* norm_k_w       = (const float*)d_in[22];
    const float* norm_added_q_w = (const float*)d_in[23];
    const float* norm_added_k_w = (const float*)d_in[24];
    const float* to_out_w     = (const float*)d_in[25]; const float* to_out_b     = (const float*)d_in[26];
    const float* to_add_out_w = (const float*)d_in[27]; const float* to_add_out_b = (const float*)d_in[28];
    const float* img_mlp_w1 = (const float*)d_in[29]; const float* img_mlp_b1 = (const float*)d_in[30];
    const float* img_mlp_w2 = (const float*)d_in[31]; const float* img_mlp_b2 = (const float*)d_in[32];
    const float* txt_mlp_w1 = (const float*)d_in[33]; const float* txt_mlp_b1 = (const float*)d_in[34];
    const float* txt_mlp_w2 = (const float*)d_in[35]; const float* txt_mlp_b2 = (const float*)d_in[36];

    char* ws = (char*)d_ws;
    size_t off = 0;
    auto alloc = [&](size_t bytes) -> void* { void* p = ws + off; off = alnup(off + bytes); return p; };

    const size_t SD  = (size_t)S_ALL * DIMX;
    const size_t SM  = (size_t)S_ALL * MLPX;
    const size_t W33 = (size_t)DIMX * DIMX;
    const size_t W3M = (size_t)DIMX * MLPX;

    float*    t_vec   = (float*)alloc(DIMX * 4);
    float*    img_mod = (float*)alloc(6 * DIMX * 4);
    float*    txt_mod = (float*)alloc(6 * DIMX * 4);
    float*    maskf   = (float*)alloc(S_ALL * 4);
    uint16_t* xn1     = (uint16_t*)alloc(SD * 2);
    uint16_t* xn2     = (uint16_t*)alloc(SD * 2);
    float*    qf      = (float*)alloc(SD * 4);
    float*    kf      = (float*)alloc(SD * 4);
    uint16_t* qb      = (uint16_t*)alloc(SD * 2);
    uint16_t* kb      = (uint16_t*)alloc(SD * 2);
    uint16_t* vb      = (uint16_t*)alloc(SD * 2);
    uint16_t* attn_bf = (uint16_t*)alloc(SD * 2);
    float*    proj    = (float*)alloc(SD * 4);
    float*    cur     = (float*)alloc(SD * 4);
    float*    mlpout  = (float*)alloc(SD * 4);
    uint16_t* h1      = (uint16_t*)alloc(SM * 2);
    uint16_t* wq_t = (uint16_t*)alloc(W33 * 2); uint16_t* wq_i = (uint16_t*)alloc(W33 * 2);
    uint16_t* wk_t = (uint16_t*)alloc(W33 * 2); uint16_t* wk_i = (uint16_t*)alloc(W33 * 2);
    uint16_t* wv_t = (uint16_t*)alloc(W33 * 2); uint16_t* wv_i = (uint16_t*)alloc(W33 * 2);
    uint16_t* wo_t = (uint16_t*)alloc(W33 * 2); uint16_t* wo_i = (uint16_t*)alloc(W33 * 2);
    uint16_t* wm1_t = (uint16_t*)alloc(W3M * 2); uint16_t* wm1_i = (uint16_t*)alloc(W3M * 2);
    uint16_t* wm2_t = (uint16_t*)alloc(W3M * 2); uint16_t* wm2_i = (uint16_t*)alloc(W3M * 2);

    // 1. modulation vectors + mask
    silu_kernel<<<12, 256, 0, stream>>>(temb, t_vec, DIMX);
    modvec_kernel<<<(2 * 6 * DIMX) / 256, 256, 0, stream>>>(
        t_vec, img_mod_w, img_mod_b, txt_mod_w, txt_mod_b, img_mod, txt_mod);
    mask_kernel<<<10, 256, 0, stream>>>(mask_in, maskf);

    // 2. weight conversion fp32 -> bf16 (per call; deterministic)
    auto cvt = [&](const float* src, uint16_t* dst, size_t n) {
        cvt_bf16_kernel<<<4096, 256, 0, stream>>>(src, dst, n);
    };
    cvt(add_q_w, wq_t, W33);      cvt(to_q_w, wq_i, W33);
    cvt(add_k_w, wk_t, W33);      cvt(to_k_w, wk_i, W33);
    cvt(add_v_w, wv_t, W33);      cvt(to_v_w, wv_i, W33);
    cvt(to_add_out_w, wo_t, W33); cvt(to_out_w, wo_i, W33);
    cvt(txt_mlp_w1, wm1_t, W3M);  cvt(img_mlp_w1, wm1_i, W3M);
    cvt(txt_mlp_w2, wm2_t, W3M);  cvt(img_mlp_w2, wm2_i, W3M);

    // 3. LN + modulate (msa) -> xn1 bf16 (txt rows first)
    ln_mod_kernel<<<S_ALL, 256, 0, stream>>>(ehs, hs, txt_mod, img_mod, xn1);

    // 4. QKV projections (WMMA GEMMs)
    auto gemm = [&](const uint16_t* A, const uint16_t* B, const float* bias,
                    float* oF, uint16_t* oH, int M, int N, int K, int mode) {
        gemm_bf16_kernel<<<dim3(N / 128, M / 128), 256, 0, stream>>>(A, B, bias, oF, oH, M, N, K, mode);
    };
    gemm(xn1,                 wq_t, add_q_b, qf,                 nullptr, S_TXT, DIMX, DIMX, 0);
    gemm(xn1 + S_TXT * DIMX,  wq_i, to_q_b,  qf + S_TXT * DIMX,  nullptr, S_IMG, DIMX, DIMX, 0);
    gemm(xn1,                 wk_t, add_k_b, kf,                 nullptr, S_TXT, DIMX, DIMX, 0);
    gemm(xn1 + S_TXT * DIMX,  wk_i, to_k_b,  kf + S_TXT * DIMX,  nullptr, S_IMG, DIMX, DIMX, 0);
    gemm(xn1,                 wv_t, add_v_b, nullptr, vb,                 S_TXT, DIMX, DIMX, 2);
    gemm(xn1 + S_TXT * DIMX,  wv_i, to_v_b,  nullptr, vb + S_TXT * DIMX,  S_IMG, DIMX, DIMX, 2);

    // 5. RMSNorm + RoPE -> bf16
    rmsrope_kernel<<<S_ALL * HEADSX, 128, 0, stream>>>(qf, norm_added_q_w, norm_q_w, freqs, qb);
    rmsrope_kernel<<<S_ALL * HEADSX, 128, 0, stream>>>(kf, norm_added_k_w, norm_k_w, freqs, kb);

    // 6. attention
    flash_attn_kernel<<<dim3(S_ALL / 128, HEADSX), 256, 0, stream>>>(qb, kb, vb, maskf, attn_bf);

    // 7. output projections + gated residual
    gemm(attn_bf,                wo_t, to_add_out_b, proj,                nullptr, S_TXT, DIMX, DIMX, 0);
    gemm(attn_bf + S_TXT * DIMX, wo_i, to_out_b,     proj + S_TXT * DIMX, nullptr, S_IMG, DIMX, DIMX, 0);
    gate_add_kernel<<<(unsigned)((SD + 255) / 256), 256, 0, stream>>>(
        ehs, hs, proj, txt_mod + 2 * DIMX, img_mod + 2 * DIMX, cur, SD);

    // 8. MLP: LN+mod2 -> gemm1(gelu->bf16) -> gemm2 -> gated residual into d_out
    ln_mod_kernel<<<S_ALL, 256, 0, stream>>>(cur, cur + S_TXT * DIMX,
                                             txt_mod + 3 * DIMX, img_mod + 3 * DIMX, xn2);
    gemm(xn2,                wm1_t, txt_mlp_b1, nullptr, h1,                 S_TXT, MLPX, DIMX, 1);
    gemm(xn2 + S_TXT * DIMX, wm1_i, img_mlp_b1, nullptr, h1 + S_TXT * MLPX,  S_IMG, MLPX, DIMX, 1);
    gemm(h1,                 wm2_t, txt_mlp_b2, mlpout,                nullptr, S_TXT, DIMX, MLPX, 0);
    gemm(h1 + S_TXT * MLPX,  wm2_i, img_mlp_b2, mlpout + S_TXT * DIMX, nullptr, S_IMG, DIMX, MLPX, 0);
    gate_add_kernel<<<(unsigned)((SD + 255) / 256), 256, 0, stream>>>(
        cur, cur + S_TXT * DIMX, mlpout, txt_mod + 5 * DIMX, img_mod + 5 * DIMX,
        (float*)d_out, SD);
}
